// NequIP_17832704213478
// MI455X (gfx1250) — matrix operand order repaired and checked by reading, not compiled, not run on hardware
//
#include <hip/hip_runtime.h>
#include <math.h>

typedef __attribute__((ext_vector_type(16))) _Float16 v16h;
typedef __attribute__((ext_vector_type(8)))  _Float16 v8h;
typedef __attribute__((ext_vector_type(8)))  float    v8f;

#define HS 72            // padded row stride (halves) for LDS tiles
#define SQ3F 1.7320508075688772f
#define PIF  3.14159265358979323846f

__device__ __forceinline__ void lds_fence_wave() {
    // CDNA5 split wait counter: wave-local LDS ordering without a workgroup barrier
    asm volatile("s_wait_dscnt 0x0" ::: "memory");
}

__device__ __forceinline__ float fast_tanh(float x) {
#if __has_builtin(__builtin_amdgcn_tanhf)
    return __builtin_amdgcn_tanhf(x);          // CDNA5 v_tanh_f32
#else
    // branch-free: tanh(x) = 1 - 2/(exp2(2x*log2 e) + 1)
    float e = __builtin_amdgcn_exp2f(x * 2.8853900817779268f);
    return 1.0f - 2.0f * __builtin_amdgcn_rcpf(e + 1.0f);
#endif
}

__device__ __forceinline__ float fast_sigmoid(float x) {
    return __builtin_amdgcn_rcpf(1.0f + __builtin_amdgcn_exp2f(-1.4426950408889634f * x));
}

__device__ __forceinline__ float gelu_f(float x) {
    float inner = 0.7978845608028654f * (x + 0.044715f * x * x * x);
    return 0.5f * x * (1.0f + fast_tanh(inner));
}

// A fragment: 16x32 f16, lane L(0-15)=row L holds K {kBase+0..7, kBase+16..23},
// lane L+16 holds K {kBase+8..15, kBase+24..31}  (per ISA 7.12.2)
__device__ __forceinline__ v16h load_fragA(const _Float16* base, int lane, int kBase) {
    const int row  = lane & 15;
    const int kOff = kBase + ((lane >> 4) << 3);
    const _Float16* p = base + row * HS + kOff;
    v8h lo = *(const v8h*)(p);
    v8h hi = *(const v8h*)(p + 16);
    return __builtin_shufflevector(lo, hi, 0,1,2,3,4,5,6,7,8,9,10,11,12,13,14,15);
}

// B fragment: 32x16 f16 (KxN), lane L(0-15)=col N holds K {0..7, 16..23}, lane L+16 holds K {8..15, 24..31}.
// Weights stored transposed in LDS: wb[n*HS + k] -> contiguous in K.
__device__ __forceinline__ v16h load_fragB(const _Float16* base, int lane, int ntile, int kBase) {
    const int n    = ntile * 16 + (lane & 15);
    const int kOff = kBase + ((lane >> 4) << 3);
    const _Float16* p = base + n * HS + kOff;
    v8h lo = *(const v8h*)(p);
    v8h hi = *(const v8h*)(p + 16);
    return __builtin_shufflevector(lo, hi, 0,1,2,3,4,5,6,7,8,9,10,11,12,13,14,15);
}

// One 64->64 MLP layer for a 16-row tile living in hb (f16, stride HS), weights wb (f16^T).
__device__ __forceinline__ void mlp_layer64(_Float16* hb, const _Float16* wb, int lane) {
    v16h a0 = load_fragA(hb, lane, 0);
    v16h a1 = load_fragA(hb, lane, 32);
    v8f acc[4];
#pragma unroll
    for (int nt = 0; nt < 4; ++nt) {
        v16h b0 = load_fragB(wb, lane, nt, 0);
        v16h b1 = load_fragB(wb, lane, nt, 32);
        v8f c = {};
        c = __builtin_amdgcn_wmma_f32_16x16x32_f16(false, a0, false, b0, (short)0, c, false, false);
        c = __builtin_amdgcn_wmma_f32_16x16x32_f16(false, a1, false, b1, (short)0, c, false, false);
        acc[nt] = c;
    }
    const int rb  = (lane >> 4) * 8;
    const int col = lane & 15;
#pragma unroll
    for (int j = 0; j < 8; ++j) {
#pragma unroll
        for (int nt = 0; nt < 4; ++nt)
            hb[(rb + j) * HS + nt * 16 + col] = (_Float16)gelu_f(acc[nt][j]);
    }
}

__global__ void __launch_bounds__(256)
edge_kernel(const float* __restrict__ nodes, const int* __restrict__ senders,
            const int* __restrict__ receivers, const float* __restrict__ w_msg,
            const float* __restrict__ W0, const float* __restrict__ W1,
            const float* __restrict__ W2, const float* __restrict__ W3,
            int t, int E, int eBlocks, float* __restrict__ Ms, float* __restrict__ Mv)
{
    __shared__ __align__(16) _Float16 sWt1[64][HS];
    __shared__ __align__(16) _Float16 sWt2[64][HS];
    __shared__ __align__(16) _Float16 sWt3[16][HS];
    __shared__ float sW0[4][64];
    __shared__ __align__(16) _Float16 sH[8][16][HS];
    __shared__ float sR[8][32][4];
    __shared__ float sMix[8][32][8];

    const int tid  = threadIdx.x;
    const int lane = tid & 31;
    const int wave = tid >> 5;

    const float* W0t = W0 + (size_t)t * 4 * 64;
    const float* W1t = W1 + (size_t)t * 64 * 64;
    const float* W2t = W2 + (size_t)t * 64 * 64;
    const float* W3t = W3 + (size_t)t * 64 * 10;

    // Stage weights once per (persistent) block: transpose + f16-convert into LDS
    for (int i = tid; i < 64 * 64; i += 256) {
        int k = i >> 6, n = i & 63;
        sWt1[n][k] = (_Float16)W1t[i];
        sWt2[n][k] = (_Float16)W2t[i];
    }
    for (int i = tid; i < 16 * 64; i += 256)
        sWt3[i >> 6][i & 63] = (_Float16)0.0f;
    sW0[tid >> 6][tid & 63] = W0t[tid];
    __syncthreads();
    for (int i = tid; i < 64 * 10; i += 256) {
        int k = i / 10, n = i - 10 * k;
        sWt3[n][k] = (_Float16)W3t[i];   // padded cols 10..15 stay zero
    }
    __syncthreads();

    const float wm = w_msg[t];
    _Float16* hb = &sH[wave][0][0];
    const int row = lane & 15;
    const int cb  = (lane >> 4) * 32;

    // Persistent grid-stride loop over 256-edge tiles (steady state is wave-local only)
    for (int blk = blockIdx.x; blk < eBlocks; blk += gridDim.x) {
        const long eIdx = (long)blk * 256 + tid;
        const bool valid = eIdx < (long)E;
        const int e  = valid ? (int)eIdx : 0;
        const int si = senders[e];
        const int ri = receivers[e];
        const float vs0 = nodes[3 * si + 0], vs1 = nodes[3 * si + 1], vs2 = nodes[3 * si + 2];
        const float vr0 = nodes[3 * ri + 0], vr1 = nodes[3 * ri + 1], vr2 = nodes[3 * ri + 2];
        const float r0 = vs0 - vr0, r1 = vs1 - vr1, r2 = vs2 - vr2;
        const float d  = sqrtf(r0 * r0 + r1 * r1 + r2 * r2);
        const float ds = fmaxf(d, 1e-9f);
        const float inv = 1.0f / ds;
        const float vm0 = wm * vs0, vm1 = wm * vs1, vm2 = wm * vs2;
        const float va0 = SQ3F * r0 * inv, va1 = SQ3F * r1 * inv, va2 = SQ3F * r2 * inv;
        const float s0 = (vm0 * va0 + vm1 * va1 + vm2 * va2) * (1.0f / SQ3F);
        const float s1 = 1.0f;
        const float s2 = (va0 * va0 + va1 * va1 + va2 * va2) * (1.0f / SQ3F);
#pragma unroll
        for (int k = 0; k < 4; ++k)
            sR[wave][lane][k] = 1.4142135623730951f * __sinf((float)(k + 1) * PIF * ds) * inv;
        lds_fence_wave();

#pragma unroll
        for (int tile = 0; tile < 2; ++tile) {
            // layer 0 (K=4) in VALU -> f16 activations in LDS tile
            const float q0 = sR[wave][tile * 16 + row][0];
            const float q1 = sR[wave][tile * 16 + row][1];
            const float q2 = sR[wave][tile * 16 + row][2];
            const float q3 = sR[wave][tile * 16 + row][3];
#pragma unroll
            for (int cc = 0; cc < 32; cc += 8) {
                v8h pk;
#pragma unroll
                for (int j = 0; j < 8; ++j) {
                    int c = cb + cc + j;
                    float a = q0 * sW0[0][c] + q1 * sW0[1][c] + q2 * sW0[2][c] + q3 * sW0[3][c];
                    pk[j] = (_Float16)gelu_f(a);
                }
                *(v8h*)&sH[wave][row][cb + cc] = pk;
            }
            lds_fence_wave();

            mlp_layer64(hb, &sWt1[0][0], lane);   // 8x v_wmma_f32_16x16x32_f16
            lds_fence_wave();
            mlp_layer64(hb, &sWt2[0][0], lane);   // 8x
            lds_fence_wave();

            // layer 3: 64 -> 10 (padded to one N-tile), 2x WMMA
            {
                v16h a0 = load_fragA(hb, lane, 0);
                v16h a1 = load_fragA(hb, lane, 32);
                v16h b0 = load_fragB(&sWt3[0][0], lane, 0, 0);
                v16h b1 = load_fragB(&sWt3[0][0], lane, 0, 32);
                v8f c = {};
                c = __builtin_amdgcn_wmma_f32_16x16x32_f16(false, a0, false, b0, (short)0, c, false, false);
                c = __builtin_amdgcn_wmma_f32_16x16x32_f16(false, a1, false, b1, (short)0, c, false, false);
                const int rb  = (lane >> 4) * 8;
                const int col = lane & 15;
                if (col < 6) {
#pragma unroll
                    for (int j = 0; j < 8; ++j)
                        sMix[wave][tile * 16 + rb + j][col] = c[j];
                }
            }
            lds_fence_wave();
        }

        // ---- apply mix, aggregate to receivers (L2-resident f32 atomics) ----
        const float m0 = sMix[wave][lane][0], m1 = sMix[wave][lane][1], m2 = sMix[wave][lane][2];
        const float m3 = sMix[wave][lane][3], m4 = sMix[wave][lane][4], m5 = sMix[wave][lane][5];
        if (valid) {
            float* MsR = Ms + 3 * (size_t)ri;
            atomicAdd(&MsR[0], s0 * m0);
            atomicAdd(&MsR[1], s1 * m1);
            atomicAdd(&MsR[2], s2 * m2);
            float* MvR = Mv + 9 * (size_t)ri;
            atomicAdd(&MvR[0], vm0 * m3); atomicAdd(&MvR[1], vm1 * m3); atomicAdd(&MvR[2], vm2 * m3);
            atomicAdd(&MvR[3], va0 * m4); atomicAdd(&MvR[4], va1 * m4); atomicAdd(&MvR[5], va2 * m4);
            atomicAdd(&MvR[6], va0 * m5); atomicAdd(&MvR[7], va1 * m5); atomicAdd(&MvR[8], va2 * m5);
        }
        lds_fence_wave();
    }
}

__global__ void zero_kernel(float* __restrict__ p, int n) {
    int i = blockIdx.x * blockDim.x + threadIdx.x;
    if (i < n) p[i] = 0.0f;
}

__global__ void deg_kernel(const int* __restrict__ receivers, float* __restrict__ deg, int E) {
    int i = blockIdx.x * blockDim.x + threadIdx.x;
    if (i < E) atomicAdd(&deg[receivers[i]], 1.0f);
}

__global__ void node_kernel(const float* __restrict__ nodes_in, const float* __restrict__ Ms,
                            const float* __restrict__ Mv, const float* __restrict__ deg,
                            const float* __restrict__ A_s, const float* __restrict__ A_v,
                            const float* __restrict__ b_node, const float* __restrict__ w_post,
                            int t, int N, float* __restrict__ nodes_out)
{
    int n = blockIdx.x * blockDim.x + threadIdx.x;
    if (n >= N) return;
    const float invd = 1.0f / fmaxf(deg[n], 1.0f);
    const float as0 = A_s[3 * t + 0], as1 = A_s[3 * t + 1], as2 = A_s[3 * t + 2];
    const float av0 = A_v[3 * t + 0], av1 = A_v[3 * t + 1], av2 = A_v[3 * t + 2];
    const float bn = b_node[t], wp = w_post[t];
    const float gate = (Ms[3 * n + 0] * as0 + Ms[3 * n + 1] * as1 + Ms[3 * n + 2] * as2) * invd;
    const float sg = fast_sigmoid(gate);
    const float* mv = Mv + 9 * (size_t)n;
#pragma unroll
    for (int c = 0; c < 3; ++c) {
        float v = (mv[0 + c] * av0 + mv[3 + c] * av1 + mv[6 + c] * av2) * invd
                + bn * nodes_in[3 * n + c];
        nodes_out[3 * n + c] = wp * sg * v;
    }
}

extern "C" void kernel_launch(void* const* d_in, const int* in_sizes, int n_in,
                              void* d_out, int out_size, void* d_ws, size_t ws_size,
                              hipStream_t stream) {
    (void)n_in; (void)out_size; (void)ws_size;
    const float* pos       = (const float*)d_in[0];
    const int*   senders   = (const int*)d_in[1];
    const int*   receivers = (const int*)d_in[2];
    const float* w_msg     = (const float*)d_in[3];
    const float* W0        = (const float*)d_in[4];
    const float* W1        = (const float*)d_in[5];
    const float* W2        = (const float*)d_in[6];
    const float* W3        = (const float*)d_in[7];
    const float* A_s       = (const float*)d_in[8];
    const float* A_v       = (const float*)d_in[9];
    const float* b_node    = (const float*)d_in[10];
    const float* w_post    = (const float*)d_in[11];

    const int N = in_sizes[0] / 3;
    const int E = in_sizes[1];
    const int S = in_sizes[3];

    float* ws     = (float*)d_ws;
    float* nodesA = ws;                 // N*3
    float* nodesB = nodesA + (size_t)N * 3;
    float* Ms     = nodesB + (size_t)N * 3;
    float* Mv     = Ms + (size_t)N * 3; // N*9
    float* deg    = Mv + (size_t)N * 9; // N

    const int eBlocks = (E + 255) / 256;
    const int nBlocks = (N + 255) / 256;
    const int pGrid   = (eBlocks < 1536) ? eBlocks : 1536;  // persistent grid

    // receivers are step-invariant -> degree computed once
    zero_kernel<<<nBlocks, 256, 0, stream>>>(deg, N);
    deg_kernel<<<(E + 255) / 256, 256, 0, stream>>>(receivers, deg, E);

    const float* cur = pos;
    for (int t = 0; t < S; ++t) {
        zero_kernel<<<(N * 3 + 255) / 256, 256, 0, stream>>>(Ms, N * 3);
        zero_kernel<<<(N * 9 + 255) / 256, 256, 0, stream>>>(Mv, N * 9);
        edge_kernel<<<pGrid, 256, 0, stream>>>(cur, senders, receivers, w_msg,
                                               W0, W1, W2, W3, t, E, eBlocks, Ms, Mv);
        float* nxt = (t == S - 1) ? (float*)d_out : ((t & 1) ? nodesB : nodesA);
        node_kernel<<<nBlocks, 256, 0, stream>>>(cur, Ms, Mv, deg, A_s, A_v,
                                                 b_node, w_post, t, N, nxt);
        cur = nxt;
    }
}
